// SparseDistributedMemory_52587579572259
// MI455X (gfx1250) — matrix operand order repaired and checked by reading, not compiled, not run on hardware
//
#include <hip/hip_runtime.h>

typedef __attribute__((ext_vector_type(16))) __bf16 v16bf;
typedef __attribute__((ext_vector_type(8)))  __bf16 v8bf;
typedef __attribute__((ext_vector_type(4)))  __bf16 v4bf;
typedef __attribute__((ext_vector_type(8)))  float  v8f;

#define B_ROWS     4096
#define IN_SIZE    256
#define MEM_SZ     65536
#define VSZ        8
#define SPARS      32
#define LR_OVER_S  (0.1f / 32.0f)

#define NCHUNKS    8
#define CHUNK_COLS (MEM_SZ / NCHUNKS)                 // 8192
#define WPB        8                                   // waves per block
#define COLS_PER_WAVE (CHUNK_COLS / WPB)               // 1024
#define NTILES     (COLS_PER_WAVE / 16)                // 64
#define MTILES     (B_ROWS / 16)                       // 256
#define NEG_INF    -3.0e38f

// ---------------------------------------------------------------------------
// Kernel 0: one-time f32 -> bf16 conversion of proj (64 MB -> 32 MB).
// ---------------------------------------------------------------------------
__global__ void sdm_cvt_proj(const float4* __restrict__ in,
                             v4bf* __restrict__ out, int n4)
{
  const int i = blockIdx.x * blockDim.x + threadIdx.x;
  if (i < n4) {
    const float4 f = in[i];
    v4bf o;
    o[0] = (__bf16)f.x; o[1] = (__bf16)f.y;
    o[2] = (__bf16)f.z; o[3] = (__bf16)f.w;
    out[i] = o;
  }
}

// ---------------------------------------------------------------------------
// Kernel 1: bf16 WMMA scoring fused with per-wave top-32 selection.
// grid = (MTILES, NCHUNKS), block = 256 (8 waves).
// Each wave: 16 rows x 1024 cols, software-pipelined B double-buffer.
// ---------------------------------------------------------------------------
__global__ __launch_bounds__(256) void sdm_score_topk(
    const float* __restrict__ keys, const __bf16* __restrict__ projh,
    float* __restrict__ cand_val, int* __restrict__ cand_idx)
{
  __shared__ float s_scores[WPB][16][17];     // per-wave 16x16 score scratch (padded)
  __shared__ float s_cval[WPB][16][SPARS];    // per-wave per-row top-32 scores
  __shared__ int   s_cidx[WPB][16][SPARS];    // per-wave per-row top-32 indices

  const int lane  = threadIdx.x & 31;
  const int wave  = threadIdx.x >> 5;
  const int mtile = blockIdx.x;
  const int chunk = blockIdx.y;

  if (lane < 16) {
    #pragma unroll
    for (int i = 0; i < SPARS; ++i) {
      s_cval[wave][lane][i] = NEG_INF;
      s_cidx[wave][lane][i] = 0;
    }
  }

  // --- Preload A fragments: 16(M) x 256(K) keys tile as bf16 (once) -------
  const int m   = lane & 15;
  const int kbA = (lane < 16) ? 0 : 16;
  const float* krow = keys + (size_t)(mtile * 16 + m) * IN_SIZE;
  v16bf afrag[8];
  #pragma unroll
  for (int kk = 0; kk < 8; ++kk) {
    const int k0 = kk * 32;
    #pragma unroll
    for (int j = 0; j < 8; ++j) {
      afrag[kk][j]     = (__bf16)krow[k0 + kbA + j];
      afrag[kk][8 + j] = (__bf16)krow[k0 + kbA + 8 + j];
    }
  }

  // --- B layout (16-bit 32x16): lane l holds column N=l%16, K = kbB..kbB+15
  const int n    = lane & 15;
  const int kbB  = (lane < 16) ? 0 : 16;
  const int col0 = chunk * CHUNK_COLS + wave * COLS_PER_WAVE;

  float mn = NEG_INF;   // running min of this lane's row-list (lanes 0..15)
  int   mp = 0;

  // Load all 8 B fragments of tile t into a register buffer (16x b128).
  auto load_tile = [&](int t, v16bf* bfrag) {
    const __bf16* prow = projh + (size_t)(col0 + t * 16 + n) * IN_SIZE + kbB;
    #pragma unroll
    for (int kk = 0; kk < 8; ++kk) {
      const v8bf b0 = *(const v8bf*)(prow + kk * 32);
      const v8bf b1 = *(const v8bf*)(prow + kk * 32 + 8);
      bfrag[kk] = __builtin_shufflevector(
          b0, b1, 0, 1, 2, 3, 4, 5, 6, 7, 8, 9, 10, 11, 12, 13, 14, 15);
    }
  };

  // 8 WMMAs + LDS scatter + top-32 insertion scan for one 16x16 tile.
  auto process_tile = [&](int t, const v16bf* bfrag) {
    v8f acc = {};
    #pragma unroll
    for (int kk = 0; kk < 8; ++kk)
      acc = __builtin_amdgcn_wmma_f32_16x16x32_bf16(
          /*neg_a=*/false, afrag[kk], /*neg_b=*/false, bfrag[kk],
          /*c_mod=*/(short)0, acc, /*reuse_a=*/false, /*reuse_b=*/false);

    const int colbase = col0 + t * 16;
    const int rbase   = (lane < 16) ? 0 : 8;
    #pragma unroll
    for (int r = 0; r < 8; ++r) s_scores[wave][rbase + r][n] = acc[r];
    asm volatile("s_wait_dscnt 0x0" ::: "memory");   // intra-wave LDS RAW

    if (lane < 16) {
      const int row = lane;
      for (int c = 0; c < 16; ++c) {
        const float sc = s_scores[wave][row][c];
        if (sc > mn) {
          s_cval[wave][row][mp] = sc;
          s_cidx[wave][row][mp] = colbase + c;
          mn = s_cval[wave][row][0]; mp = 0;
          #pragma unroll 8
          for (int i = 1; i < SPARS; ++i) {
            const float v = s_cval[wave][row][i];
            if (v < mn) { mn = v; mp = i; }
          }
        }
      }
    }
  };

  // Software pipeline: loads for tile t+1 fly behind tile t's WMMAs + top-k.
  v16bf bufA[8], bufB[8];
  load_tile(0, bufA);
  for (int t = 0; t < NTILES; t += 2) {
    load_tile(t + 1, bufB);            // NTILES is even
    process_tile(t, bufA);
    if (t + 2 < NTILES) load_tile(t + 2, bufA);
    process_tile(t + 1, bufB);
  }

  __syncthreads();

  // --- Block merge: wave w selects top-32 of 8x32 candidates for rows 2w,2w+1
  for (int rr = 0; rr < 2; ++rr) {
    const int row = wave * 2 + rr;
    float cv[8]; int ci[8];
    #pragma unroll
    for (int w2 = 0; w2 < 8; ++w2) {
      cv[w2] = s_cval[w2][row][lane];
      ci[w2] = s_cidx[w2][row][lane];
    }
    unsigned taken = 0;
    const size_t obase = (((size_t)mtile * NCHUNKS + chunk) * 16 + row) * SPARS;
    for (int round = 0; round < SPARS; ++round) {
      float lm = NEG_INF; int ls = -1; int li = 0;
      #pragma unroll
      for (int i = 0; i < 8; ++i)
        if (!((taken >> i) & 1u) && cv[i] > lm) { lm = cv[i]; ls = i; li = ci[i]; }
      float rv = lm; int rl = lane; int rs = ls;
      #pragma unroll
      for (int off = 16; off > 0; off >>= 1) {
        const float ov = __shfl_xor(rv, off);
        const int   ol = __shfl_xor(rl, off);
        const int   os = __shfl_xor(rs, off);
        if (ov > rv || (ov == rv && ol < rl)) { rv = ov; rl = ol; rs = os; }
      }
      const int widx = __shfl(li, rl);                // winner's index to all lanes
      if (lane == rl && rs >= 0) taken |= (1u << rs);
      if (lane == round) { cand_val[obase + round] = rv; cand_idx[obase + round] = widx; }
    }
  }
}

// ---------------------------------------------------------------------------
// Kernel 2: one wave per batch row. Merge 8 chunk-lists (256 cands) -> top-32
// in descending order, then gather-sum retrieval + delta computation.
// ---------------------------------------------------------------------------
__global__ __launch_bounds__(256) void sdm_merge_retrieve(
    const float* __restrict__ cand_val, const int* __restrict__ cand_idx,
    const float* __restrict__ mem_value, const float* __restrict__ targets,
    float* __restrict__ out_retrieved, int* __restrict__ sel_idx,
    float* __restrict__ deltas)
{
  const int lane  = threadIdx.x & 31;
  const int wave  = threadIdx.x >> 5;
  const int row   = blockIdx.x * WPB + wave;
  const int mtile = row >> 4;
  const int r     = row & 15;

  float cv[8]; int ci[8];
  #pragma unroll
  for (int c = 0; c < 8; ++c) {
    const size_t o = (((size_t)mtile * NCHUNKS + c) * 16 + r) * SPARS + lane;
    cv[c] = cand_val[o]; ci[c] = cand_idx[o];
  }

  unsigned taken = 0;
  int mysel = 0;                                     // lane k keeps the k-th pick
  for (int round = 0; round < SPARS; ++round) {
    float lm = NEG_INF; int ls = -1; int li = 0;
    #pragma unroll
    for (int i = 0; i < 8; ++i)
      if (!((taken >> i) & 1u) && cv[i] > lm) { lm = cv[i]; ls = i; li = ci[i]; }
    float rv = lm; int rl = lane; int rs = ls;
    #pragma unroll
    for (int off = 16; off > 0; off >>= 1) {
      const float ov = __shfl_xor(rv, off);
      const int   ol = __shfl_xor(rl, off);
      const int   os = __shfl_xor(rs, off);
      if (ov > rv || (ov == rv && ol < rl)) { rv = ov; rl = ol; rs = os; }
    }
    const int widx = __shfl(li, rl);
    if (lane == rl && rs >= 0) taken |= (1u << rs);
    if (lane == round) mysel = widx;
  }
  sel_idx[(size_t)row * SPARS + lane] = mysel;

  // Gather-sum retrieval: lane s loads mem_value[sel_s][0..7], reduce over lanes.
  const float4* mv = (const float4*)(mem_value + (size_t)mysel * VSZ);
  const float4 a = mv[0], b = mv[1];
  float v[8] = {a.x, a.y, a.z, a.w, b.x, b.y, b.z, b.w};
  #pragma unroll
  for (int j = 0; j < 8; ++j) {
    #pragma unroll
    for (int off = 16; off > 0; off >>= 1) v[j] += __shfl_xor(v[j], off);
  }

  if (lane == 0) {
    float4* ro = (float4*)(out_retrieved + (size_t)row * VSZ);
    float4 r0 = {v[0], v[1], v[2], v[3]};
    float4 r1 = {v[4], v[5], v[6], v[7]};
    ro[0] = r0; ro[1] = r1;
    const float4* tg = (const float4*)(targets + (size_t)row * VSZ);
    const float4 t0 = tg[0], t1 = tg[1];
    float4 d0 = {(t0.x - v[0]) * LR_OVER_S, (t0.y - v[1]) * LR_OVER_S,
                 (t0.z - v[2]) * LR_OVER_S, (t0.w - v[3]) * LR_OVER_S};
    float4 d1 = {(t1.x - v[4]) * LR_OVER_S, (t1.y - v[5]) * LR_OVER_S,
                 (t1.z - v[6]) * LR_OVER_S, (t1.w - v[7]) * LR_OVER_S};
    float4* dd = (float4*)(deltas + (size_t)row * VSZ);
    dd[0] = d0; dd[1] = d1;
  }
}

// ---------------------------------------------------------------------------
// Kernel 3a: vectorized copy mem_value -> out[B..B+MEM)
// ---------------------------------------------------------------------------
__global__ void sdm_copy_mem(const float* __restrict__ mem_value,
                             float* __restrict__ out_mem, int n4)
{
  const int i = blockIdx.x * blockDim.x + threadIdx.x;
  if (i < n4) ((float4*)out_mem)[i] = ((const float4*)mem_value)[i];
}

// ---------------------------------------------------------------------------
// Kernel 3b: scatter-add deltas into out memory region (duplicate-safe atomics)
// ---------------------------------------------------------------------------
__global__ void sdm_scatter(const int* __restrict__ sel_idx,
                            const float* __restrict__ deltas,
                            float* __restrict__ out_mem)
{
  const int t = blockIdx.x * blockDim.x + threadIdx.x;   // 0 .. B*SPARS-1
  if (t >= B_ROWS * SPARS) return;
  const int b = t >> 5;
  const int s = t & 31;
  const int idx = sel_idx[(size_t)b * SPARS + s];
  const float* d = deltas + (size_t)b * VSZ;
  float* o = out_mem + (size_t)idx * VSZ;
  #pragma unroll
  for (int vv = 0; vv < VSZ; ++vv) atomicAdd(o + vv, d[vv]);
}

// ---------------------------------------------------------------------------
extern "C" void kernel_launch(void* const* d_in, const int* in_sizes, int n_in,
                              void* d_out, int out_size, void* d_ws, size_t ws_size,
                              hipStream_t stream)
{
  const float* keys      = (const float*)d_in[0];   // [B, 256]
  const float* targets   = (const float*)d_in[1];   // [B, 8]
  const float* proj      = (const float*)d_in[2];   // [65536, 256]
  const float* mem_value = (const float*)d_in[3];   // [65536, 8]

  float* out           = (float*)d_out;
  float* out_retrieved = out;                              // [B, 8]
  float* out_mem       = out + (size_t)B_ROWS * VSZ;       // [65536, 8]

  // Workspace layout (~41 MB total)
  char* ws = (char*)d_ws;
  const size_t n_cand = (size_t)MTILES * NCHUNKS * 16 * SPARS;  // 1,048,576
  float*  cand_val = (float*)ws;  ws += n_cand * sizeof(float);
  int*    cand_idx = (int*)ws;    ws += n_cand * sizeof(int);
  int*    sel_idx  = (int*)ws;    ws += (size_t)B_ROWS * SPARS * sizeof(int);
  float*  deltas   = (float*)ws;  ws += (size_t)B_ROWS * VSZ * sizeof(float);
  __bf16* projh    = (__bf16*)ws;                        // [65536, 256] bf16

  // 0) pre-convert proj to bf16 (once; 64 MB read + 32 MB write)
  const int np4 = MEM_SZ * IN_SIZE / 4;                  // 4,194,304 float4s
  sdm_cvt_proj<<<(np4 + 255) / 256, 256, 0, stream>>>(
      (const float4*)proj, (v4bf*)projh, np4);

  // 1) WMMA scoring + fused top-k
  dim3 g1(MTILES, NCHUNKS);
  sdm_score_topk<<<g1, 256, 0, stream>>>(keys, projh, cand_val, cand_idx);

  // 2) chunk merge + retrieval + deltas
  sdm_merge_retrieve<<<B_ROWS / WPB, 256, 0, stream>>>(
      cand_val, cand_idx, mem_value, targets, out_retrieved, sel_idx, deltas);

  // 3) memory copy + scatter-add update
  const int n4 = MEM_SZ * VSZ / 4;                       // 131072 float4s
  sdm_copy_mem<<<(n4 + 255) / 256, 256, 0, stream>>>(mem_value, out_mem, n4);

  sdm_scatter<<<(B_ROWS * SPARS + 255) / 256, 256, 0, stream>>>(
      sel_idx, deltas, out_mem);
}
